// GINE_Model_82721070121719
// MI455X (gfx1250) — matrix operand order repaired and checked by reading, not compiled, not run on hardware
//
#include <hip/hip_runtime.h>
#include <hip/hip_bf16.h>
#include <math.h>

typedef __attribute__((ext_vector_type(16))) _Float16 v16h;
typedef __attribute__((ext_vector_type(8)))  float    v8f;

#define NNODES 100000
#define NGRAPH 512
#define BN_EPS 1e-5f

// ---------------------------------------------------------------------------
// zero helper
// ---------------------------------------------------------------------------
__global__ void zero_kernel(float* __restrict__ p, int n) {
    int gid = blockIdx.x * blockDim.x + threadIdx.x;
    if (gid < n) p[gid] = 0.0f;
}

// ---------------------------------------------------------------------------
// Input encoding:  h0 = [z_emb | x | (rw+c2s+samehop)/3]  (35),  h = h0 @ W^T + b
// Writes both H0 and ACC (= layer-0 accumulator init).
// block = 256 = 4 nodes x 64 output channels
// ---------------------------------------------------------------------------
__global__ __launch_bounds__(256) void encode_kernel(
    const float* __restrict__ x, const float* __restrict__ rw,
    const float* __restrict__ c2s, const float* __restrict__ sh,
    const float* __restrict__ ztab, const int* __restrict__ z,
    const float* __restrict__ initW, const float* __restrict__ initb,
    float* __restrict__ H0, float* __restrict__ ACC)
{
    __shared__ float sW[64 * 35];
    __shared__ float sh0[4][36];          // +1 pad vs bank conflicts
    int t = threadIdx.x;
    for (int i = t; i < 64 * 35; i += 256) sW[i] = initW[i];

    int nodeLocal = t >> 6;               // 0..3
    int j         = t & 63;               // output channel
    int node      = blockIdx.x * 4 + nodeLocal;
    __syncthreads();

    if (node < NNODES && j < 35) {
        float v;
        if (j < 8)        v = ztab[z[node] * 8 + j];
        else if (j < 19)  v = x[node * 11 + (j - 8)];
        else { int k = j - 19;
               v = (rw[node*16+k] + c2s[node*16+k] + sh[node*16+k]) * (1.0f/3.0f); }
        sh0[nodeLocal][j] = v;
    }
    __syncthreads();

    if (node < NNODES) {
        float acc = initb[j];
        #pragma unroll
        for (int k = 0; k < 35; ++k)
            acc = fmaf(sW[j * 35 + k], sh0[nodeLocal][k], acc);
        H0 [node * 64 + j] = acc;
        ACC[node * 64 + j] = acc;
    }
}

// ---------------------------------------------------------------------------
// Scatter-add one hop:  acc[dst] += feat[src] (+ edge_attr@edge_W^T if fuseEA)
// One thread per (edge, channel): coalesced 64-float spans, f32 atomics to L2.
// Edge embedding (4->64) recomputed on the fly: avoids a 205 MB (> L2) tensor.
// ---------------------------------------------------------------------------
__global__ void scatter_kernel(
    const float* __restrict__ feat, const int* __restrict__ hop, int E,
    float* __restrict__ acc,
    const float* __restrict__ edge_attr, const float* __restrict__ edgeW, int fuseEA)
{
    int gid = blockIdx.x * blockDim.x + threadIdx.x;
    if (gid >= E * 64) return;
    int e = gid >> 6;
    int j = gid & 63;
    int s = hop[e];          // row 0: src
    int d = hop[E + e];      // row 1: dst
    float v = feat[s * 64 + j];
    if (fuseEA) {
        const float* ea = edge_attr + e * 4;
        const float* w  = edgeW + j * 4;
        v += ea[0]*w[0] + ea[1]*w[1] + ea[2]*w[2] + ea[3]*w[3];
    }
    atomicAdd(acc + d * 64 + j, v);
}

// ---------------------------------------------------------------------------
// Linear 64x64 via v_wmma_f32_16x16x32_f16 (K=64 as two chained WMMAs),
// optional fused input transform (BN-affine + ReLU from previous BN),
// fused per-channel sum / sum-of-squares accumulation for the following BN
// (LDS ds_add_f32 reduction, then 2 global atomics per channel per block).
//
// Tile = 16 nodes x 16 out-channels per wave.  N=100000 -> 6250 row-tiles,
// 25000 tiles total, 8 waves/block -> grid = 3125 (exact, no remainder).
//
// VGPR layouts per CDNA5 ISA §7.12.2 (wave32):
//   A 16x32 f16 : row m = lane&15 ; half=lane>>4 ; vgpr v holds K pair at
//                 kbase = (v&3)*2 + (v>>2)*16 + half*8
//   B 32x16 f16 : col j = lane&15 ; vgpr v holds K pair at the same kbase as A
//   C/D 16x16 f32: col j = lane&15 ; vgpr i holds row m = 8*half + i
// ---------------------------------------------------------------------------
__global__ __launch_bounds__(256) void lin_wmma_kernel(
    const float* __restrict__ in, const float* __restrict__ W,
    const float* __restrict__ bias,
    const float* __restrict__ preParams,   // [0:64)=scale  [64:128)=shift ; may be null
    int preRelu,
    float* __restrict__ out,
    float* __restrict__ statSum, float* __restrict__ statSq)
{
    __shared__ float sds[128];
    int t = threadIdx.x;
    if (t < 128) sds[t] = 0.0f;
    __syncthreads();

    int wave = t >> 5;
    int lane = t & 31;
    int tile = blockIdx.x * 8 + wave;
    const int nTiles = (NNODES / 16) * 4;

    float s = 0.0f, q = 0.0f;
    int jcol = 0;
    bool active = (tile < nTiles);

    if (active) {
        int tm   = tile >> 2;             // node tile
        int tj   = tile & 3;              // channel tile
        int m    = lane & 15;
        int half = lane >> 4;
        int node = tm * 16 + m;
        jcol     = tj * 16 + m;

        // ---- load A (16x64 strip, two K=32 chunks), optional BN+ReLU fuse
        v16h a0, a1;
        const float* rowp = in + node * 64;
        #pragma unroll
        for (int v = 0; v < 8; ++v) {
            int kb = ((v & 3) << 1) + ((v >> 2) << 4) + (half << 3);
            float x0 = rowp[kb],      x1 = rowp[kb + 1];
            float y0 = rowp[32 + kb], y1 = rowp[33 + kb];
            if (preParams) {
                x0 = fmaf(preParams[kb],      x0, preParams[64 + kb]);
                x1 = fmaf(preParams[kb + 1],  x1, preParams[65 + kb]);
                y0 = fmaf(preParams[32 + kb], y0, preParams[96 + kb]);
                y1 = fmaf(preParams[33 + kb], y1, preParams[97 + kb]);
                if (preRelu) {
                    x0 = fmaxf(x0, 0.0f); x1 = fmaxf(x1, 0.0f);
                    y0 = fmaxf(y0, 0.0f); y1 = fmaxf(y1, 0.0f);
                }
            }
            a0[2*v] = (_Float16)x0; a0[2*v+1] = (_Float16)x1;
            a1[2*v] = (_Float16)y0; a1[2*v+1] = (_Float16)y1;
        }

        // ---- load B = W^T tile (B[k][j] = W[j][k])
        v16h b0, b1;
        const float* wcol = W + jcol * 64;
        #pragma unroll
        for (int v = 0; v < 8; ++v) {
            int k = ((v & 3) << 1) + ((v >> 2) << 4) + (half << 3);
            b0[2*v] = (_Float16)wcol[k];      b0[2*v+1] = (_Float16)wcol[k + 1];
            b1[2*v] = (_Float16)wcol[32 + k]; b1[2*v+1] = (_Float16)wcol[33 + k];
        }

        v8f c = {};
        c = __builtin_amdgcn_wmma_f32_16x16x32_f16(false, a0, false, b0,
                                                   (short)0, c, false, false);
        c = __builtin_amdgcn_wmma_f32_16x16x32_f16(false, a1, false, b1,
                                                   (short)0, c, false, false);

        float bj = bias[jcol];
        #pragma unroll
        for (int i = 0; i < 8; ++i) {
            float v = c[i] + bj;
            out[(tm * 16 + (half << 3) + i) * 64 + jcol] = v;
            s += v;
            q = fmaf(v, v, q);
        }
        atomicAdd(&sds[jcol], s);        // LDS ds_add_f32
        atomicAdd(&sds[64 + jcol], q);
    }
    __syncthreads();
    if (t < 64) {
        atomicAdd(statSum + t, sds[t]);
        atomicAdd(statSq  + t, sds[64 + t]);
    }
}

// ---------------------------------------------------------------------------
// BN finalize: params = (scale, shift) from accumulated (sum, sumsq),
// then self-clean the stats buffer for its next use (stream-ordered).
// launch <<<1,64>>>
// ---------------------------------------------------------------------------
__global__ void bn_finalize_kernel(
    float* __restrict__ stats, const float* __restrict__ gamma,
    const float* __restrict__ beta, float* __restrict__ params)
{
    int j = threadIdx.x;
    const float invN = 1.0f / (float)NNODES;
    float mean = stats[j] * invN;
    float var  = stats[64 + j] * invN - mean * mean;
    float a    = gamma[j] * rsqrtf(var + BN_EPS);
    params[j]      = a;
    params[64 + j] = beta[j] - a * mean;
    stats[j] = 0.0f;
    stats[64 + j] = 0.0f;
}

// ---------------------------------------------------------------------------
// Elementwise: t = gelu?(relu(bn2(r2))), fused BN3-stat accumulation.
// ---------------------------------------------------------------------------
__global__ __launch_bounds__(256) void act_kernel(
    const float* __restrict__ r2, const float* __restrict__ params, int doGelu,
    float* __restrict__ t_out, float* __restrict__ statSum, float* __restrict__ statSq)
{
    __shared__ float sds[128];
    int t = threadIdx.x;
    if (t < 128) sds[t] = 0.0f;
    __syncthreads();

    int gid = blockIdx.x * 256 + t;
    int j   = gid & 63;
    float v = 0.0f;
    if (gid < NNODES * 64) {
        float xv = fmaf(params[j], r2[gid], params[64 + j]);
        xv = fmaxf(xv, 0.0f);
        if (doGelu)  // exact gelu (approximate=False): 0.5*x*(1+erf(x/sqrt2))
            xv = 0.5f * xv * (1.0f + erff(xv * 0.70710678118654752f));
        t_out[gid] = xv;
        v = xv;
        atomicAdd(&sds[j], v);
        atomicAdd(&sds[64 + j], v * v);
    }
    __syncthreads();
    if (t < 64) {
        atomicAdd(statSum + t, sds[t]);
        atomicAdd(statSq  + t, sds[64 + t]);
    }
}

// ---------------------------------------------------------------------------
// Apply BN3 affine -> layer output H; also seed next layer's accumulator.
// ---------------------------------------------------------------------------
__global__ void applybn_kernel(
    const float* __restrict__ t_in, const float* __restrict__ params,
    float* __restrict__ Hout, float* __restrict__ accNext)
{
    int gid = blockIdx.x * blockDim.x + threadIdx.x;
    if (gid >= NNODES * 64) return;
    int j = gid & 63;
    float v = fmaf(params[j], t_in[gid], params[64 + j]);
    Hout[gid] = v;
    if (accNext) accNext[gid] = v;
}

// ---------------------------------------------------------------------------
// Sum pooling per graph (batch is sorted; atomics into 512x64 L2-resident out)
// ---------------------------------------------------------------------------
__global__ void pool_kernel(const float* __restrict__ H,
                            const int* __restrict__ batch,
                            float* __restrict__ out)
{
    int gid = blockIdx.x * blockDim.x + threadIdx.x;
    if (gid >= NNODES * 64) return;
    int n = gid >> 6, j = gid & 63;
    atomicAdd(out + batch[n] * 64 + j, H[gid]);
}

// ---------------------------------------------------------------------------
extern "C" void kernel_launch(void* const* d_in, const int* in_sizes, int n_in,
                              void* d_out, int out_size, void* d_ws, size_t ws_size,
                              hipStream_t stream)
{
    const float* x      = (const float*)d_in[0];
    const float* rw     = (const float*)d_in[1];
    const float* c2s    = (const float*)d_in[2];
    const float* sh     = (const float*)d_in[3];
    const float* eattr  = (const float*)d_in[4];
    const float* ztab   = (const float*)d_in[5];
    const float* initW  = (const float*)d_in[6];
    const float* initb  = (const float*)d_in[7];
    const float* edgeW  = (const float*)d_in[8];
    const float* W1     = (const float*)d_in[9];
    const float* b1     = (const float*)d_in[10];
    const float* g1     = (const float*)d_in[11];
    const float* be1    = (const float*)d_in[12];
    const float* W2     = (const float*)d_in[13];
    const float* b2     = (const float*)d_in[14];
    const float* g2     = (const float*)d_in[15];
    const float* be2    = (const float*)d_in[16];
    const float* gn     = (const float*)d_in[17];
    const float* gb     = (const float*)d_in[18];
    const int*   z      = (const int*)d_in[19];
    const int*   batch  = (const int*)d_in[20];
    const int*   hops[3] = { (const int*)d_in[21], (const int*)d_in[22],
                             (const int*)d_in[23] };
    int Eh[3] = { in_sizes[21] / 2, in_sizes[22] / 2, in_sizes[23] / 2 };

    float* out = (float*)d_out;

    // ---- workspace layout (floats). Working set ~180 MB; L2-resident slices.
    const size_t NF = (size_t)NNODES * 64;
    float* ws = (float*)d_ws;
    float* H[4]  = { ws, ws + NF, ws + 2*NF, ws + 3*NF };
    float* ACC   = ws + 4 * NF;
    float* R1    = ws + 5 * NF;   // also reused as 't' post-activation buffer
    float* R2    = ws + 6 * NF;
    float* stats = ws + 7 * NF;   // 128 floats (sum | sumsq)
    float* P1    = stats + 128;   // BN1 (scale|shift)
    float* P2    = P1 + 128;      // BN2
    float* P3    = P2 + 128;      // BN3
    (void)ws_size; (void)n_in;

    const int ewBlocks  = (NNODES * 64 + 255) / 256;   // 25000
    const int linBlocks = ((NNODES / 16) * 4 + 7) / 8; // 3125

    // ws is poisoned: clear the accumulation targets we own.
    zero_kernel<<<1, 128, 0, stream>>>(stats, 128);
    zero_kernel<<<(NGRAPH * 64 + 255) / 256, 256, 0, stream>>>(out, NGRAPH * 64);

    // ---- input encoding -> H0 and ACC(layer 0)
    encode_kernel<<<(NNODES + 3) / 4, 256, 0, stream>>>(
        x, rw, c2s, sh, ztab, z, initW, initb, H[0], ACC);

    for (int L = 0; L < 3; ++L) {
        // message passing: acc += sum_i segment_sum(XX[i][src] (+ea if i==0))
        // XX[i] at layer L is H[L-i]
        for (int i = 0; i <= L; ++i) {
            int E = Eh[i];
            scatter_kernel<<<(E * 64 + 255) / 256, 256, 0, stream>>>(
                H[L - i], hops[i], E, ACC, eattr, edgeW, (i == 0) ? 1 : 0);
        }
        // Linear1 (+BN1 stats)
        lin_wmma_kernel<<<linBlocks, 256, 0, stream>>>(
            ACC, W1 + L * 4096, b1 + L * 64, nullptr, 0, R1, stats, stats + 64);
        bn_finalize_kernel<<<1, 64, 0, stream>>>(stats, g1 + L * 64, be1 + L * 64, P1);
        // BN1+ReLU fused into Linear2 operand load (+BN2 stats)
        lin_wmma_kernel<<<linBlocks, 256, 0, stream>>>(
            R1, W2 + L * 4096, b2 + L * 64, P1, 1, R2, stats, stats + 64);
        bn_finalize_kernel<<<1, 64, 0, stream>>>(stats, g2 + L * 64, be2 + L * 64, P2);
        // t = gelu?(relu(bn2(r2)))  (+BN3 stats); reuse R1 as 't'
        act_kernel<<<ewBlocks, 256, 0, stream>>>(
            R2, P2, (L < 2) ? 1 : 0, R1, stats, stats + 64);
        bn_finalize_kernel<<<1, 64, 0, stream>>>(stats, gn + L * 64, gb + L * 64, P3);
        // H[L+1] = bn3(t); also seeds next layer's accumulator
        applybn_kernel<<<ewBlocks, 256, 0, stream>>>(
            R1, P3, H[L + 1], (L < 2) ? ACC : nullptr);
    }

    // JK='last' sum pooling
    pool_kernel<<<ewBlocks, 256, 0, stream>>>(H[3], batch, out);
}
